// MultiHeadAttention_59227599012359
// MI455X (gfx1250) — matrix operand order repaired and checked by reading, not compiled, not run on hardware
//
#include <hip/hip_runtime.h>
#include <cstddef>
#include <cstdint>

// ---------------------------------------------------------------------------
// Problem constants (from reference): B=4, S=2048, D=1024, H=16, DK=64
// ---------------------------------------------------------------------------
constexpr int Bc = 4;
constexpr int Sc = 2048;
constexpr int Dc = 1024;
constexpr int Hc = 16;
constexpr int DKc = 64;
constexpr int Mrows = Bc * Sc;            // 8192 rows for all GEMMs
constexpr float SCALE = 0.125f;           // 1/sqrt(64)

typedef __bf16 bf16x16 __attribute__((ext_vector_type(16)));
typedef float f32x8 __attribute__((ext_vector_type(8)));

union FragAB {
    unsigned int u[8];
    bf16x16 v;
};

// ---------------------------------------------------------------------------
// CDNA5 async global->LDS copy (ASYNCcnt) with compile-time feature guard.
// Builtin signature (from hipcc diagnostic): param0 = int4(vector_size) ptr in
// AS1 (global), param1 = int4 ptr in AS3 (LDS), then imm offset, imm cpol.
// Falls back to a plain uint4 round-trip copy when the builtins are absent.
// ---------------------------------------------------------------------------
#if defined(__AMDGCN__) && __has_builtin(__builtin_amdgcn_global_load_async_to_lds_b128)
#define HAS_ASYNC_LDS 1
typedef int vi4 __attribute__((vector_size(16)));
static __device__ __forceinline__ void copy16_to_lds(const void* g, void* l) {
    __builtin_amdgcn_global_load_async_to_lds_b128(
        (__attribute__((address_space(1))) vi4*)g,
        (__attribute__((address_space(3))) vi4*)l, 0, 0);
}
#if __has_builtin(__builtin_amdgcn_s_wait_asynccnt)
#define WAIT_ASYNC(n) __builtin_amdgcn_s_wait_asynccnt(n)
#else
#define WAIT_ASYNC(n) asm volatile("s_wait_asynccnt %0" ::"n"(n))
#endif
#else
#define HAS_ASYNC_LDS 0
static __device__ __forceinline__ void copy16_to_lds(const void* g, void* l) {
    *(uint4*)l = *(const uint4*)g;
}
#define WAIT_ASYNC(n) ((void)0)
#endif

static __device__ __forceinline__ unsigned short f2bf(float x) {
    unsigned int u = __float_as_uint(x);
    unsigned int r = u + 0x7FFFu + ((u >> 16) & 1u);   // round-to-nearest-even
    return (unsigned short)(r >> 16);
}

// A-fragment K index for 16-bit A 16x32 layout (lanes 0-15: K 0-7,16-23;
// lanes 16-31: K 8-15,24-31; packed pairs per VGPR).
static __device__ __forceinline__ int a_k(int j, int lane) {
    int kh = (lane & 16) ? 8 : 0;
    return (j < 4) ? (kh + 2 * j) : (16 + kh + 2 * (j - 4));
}

// ---------------------------------------------------------------------------
// fp32 -> bf16 conversion (grid-stride)
// ---------------------------------------------------------------------------
__global__ void cvt_f32_bf16(const float* __restrict__ src,
                             unsigned short* __restrict__ dst, int n) {
    for (int i = blockIdx.x * blockDim.x + threadIdx.x; i < n;
         i += gridDim.x * blockDim.x)
        dst[i] = f2bf(src[i]);
}

// ---------------------------------------------------------------------------
// Tiled GEMM: C[M,1024] = A[M,1024](bf16) @ W[1024,1024](bf16) + bias
// Block: 256 threads (8 waves), tile 128x128, K-step 32, double-buffered LDS.
// A tile staged via async global->LDS DMA; W tile staged transposed manually.
// Wave grid 2x4 -> each wave computes 64x32 (4x2 subtiles of 16x16).
// ---------------------------------------------------------------------------
template <bool BF16OUT>
__global__ __launch_bounds__(256)
void gemm_bias_kernel(const unsigned short* __restrict__ A,
                      const unsigned short* __restrict__ W,
                      const float* __restrict__ bias,
                      void* __restrict__ Cout) {
    constexpr int PA = 40;   // LDS pitch for A tile [128][32] (padded)
    constexpr int PB = 40;   // LDS pitch for Bt tile [128][32] (transposed)
    __shared__ unsigned short As[2][128 * PA];
    __shared__ unsigned short Bs[2][128 * PB];

    const int tid = threadIdx.x;
    const int lane = tid & 31;
    const int w = tid >> 5;          // 0..7
    const int wm = w >> 2;           // 0..1
    const int wn = w & 3;            // 0..3

    const int mt = blockIdx.x >> 3;  // 64 M-tiles
    const int nt = blockIdx.x & 7;   // 8  N-tiles
    const int m0 = mt * 128;
    const int n0 = nt * 128;

    f32x8 acc[4][2];
    const f32x8 zf = {0.f, 0.f, 0.f, 0.f, 0.f, 0.f, 0.f, 0.f};
#pragma unroll
    for (int i = 0; i < 4; ++i)
#pragma unroll
        for (int j = 0; j < 2; ++j) acc[i][j] = zf;

    const int r = lane & 15;
    const int khB = (lane & 16) ? 16 : 0;

    // ---- tile staging: A direct (async-capable), W transposed (manual) ----
    auto stage = [&](int k0, int buf) {
        unsigned short* As_ = As[buf];
        unsigned short* Bs_ = Bs[buf];
#pragma unroll
        for (int i = 0; i < 2; ++i) {        // A: 512 chunks of 8 elems
            int c = tid + i * 256;
            int row = c >> 2;
            int kc = (c & 3) * 8;
            copy16_to_lds(&A[(size_t)(m0 + row) * Dc + k0 + kc],
                          &As_[row * PA + kc]);
        }
#pragma unroll
        for (int i = 0; i < 2; ++i) {        // W: 512 chunks of 8 elems
            int c = tid + i * 256;
            int k = c >> 4;
            int nc = (c & 15) * 8;
            uint4 d = *(const uint4*)&W[(size_t)(k0 + k) * Dc + n0 + nc];
            const unsigned short* e = (const unsigned short*)&d;
#pragma unroll
            for (int q = 0; q < 8; ++q) Bs_[(nc + q) * PB + k] = e[q];
        }
    };

    stage(0, 0);                             // prologue: tile 0 in flight
    constexpr int KT = Dc / 32;              // 32 K-tiles

    for (int kt = 0; kt < KT; ++kt) {
        const int cur = kt & 1;
        if (kt + 1 < KT) {
            stage((kt + 1) * 32, cur ^ 1);   // issue next tile (other buffer)
            WAIT_ASYNC(2);                   // tile kt's 2 async ops done
        } else {
            WAIT_ASYNC(0);
        }
        __syncthreads();                     // publish tile kt block-wide

        const unsigned short* As_ = As[cur];
        const unsigned short* Bs_ = Bs[cur];

        FragAB afr[4];
#pragma unroll
        for (int mi = 0; mi < 4; ++mi) {
            const unsigned short* arow = &As_[(wm * 64 + mi * 16 + r) * PA];
#pragma unroll
            for (int j = 0; j < 8; ++j)
                afr[mi].u[j] = *(const unsigned int*)&arow[a_k(j, lane)];
        }
        FragAB bfr[2];
#pragma unroll
        for (int ni = 0; ni < 2; ++ni) {
            const unsigned short* brow = &Bs_[(wn * 32 + ni * 16 + r) * PB];
#pragma unroll
            for (int j = 0; j < 8; ++j)
                bfr[ni].u[j] = *(const unsigned int*)&brow[a_k(j, lane)];
        }
#pragma unroll
        for (int mi = 0; mi < 4; ++mi)
#pragma unroll
            for (int ni = 0; ni < 2; ++ni)
                acc[mi][ni] = __builtin_amdgcn_wmma_f32_16x16x32_bf16(
                    false, afr[mi].v, false, bfr[ni].v, (short)0, acc[mi][ni],
                    false, false);

        __syncthreads();                     // all reads done; buffer reusable
    }

    // ---- epilogue: add bias, store ----
#pragma unroll
    for (int ni = 0; ni < 2; ++ni) {
        const int ncol = n0 + wn * 32 + ni * 16 + r;
        const float bv = bias[ncol];
#pragma unroll
        for (int mi = 0; mi < 4; ++mi) {
#pragma unroll
            for (int rr = 0; rr < 8; ++rr) {
                int Mr = wm * 64 + mi * 16 + rr + ((lane & 16) ? 8 : 0);
                float v = acc[mi][ni][rr] + bv;
                size_t idx = (size_t)(m0 + Mr) * Dc + ncol;
                if constexpr (BF16OUT)
                    ((unsigned short*)Cout)[idx] = f2bf(v);
                else
                    ((float*)Cout)[idx] = v;
            }
        }
    }
}

// ---------------------------------------------------------------------------
// Flash attention: one block = (b, h, 64-query tile); 4 waves x 16 rows.
// K tile staged row-major via async DMA (double-buffered); V tile staged
// transposed manually; P round-trips a per-wave LDS strip (C->A relayout).
// ---------------------------------------------------------------------------
__global__ __launch_bounds__(128)
void flash_attn_kernel(const unsigned short* __restrict__ Q,
                       const unsigned short* __restrict__ K,
                       const unsigned short* __restrict__ V,
                       unsigned short* __restrict__ ctx) {
    constexpr int PK = 72;                      // pitch for 64-wide tiles
    __shared__ unsigned short Kt[2][64 * PK];   // [key][dk]
    __shared__ unsigned short Vt[2][64 * PK];   // [dk][key] (transposed)
    __shared__ unsigned short Pt[4 * 16 * PK];  // per-wave [row][key]

    const int tid = threadIdx.x;
    const int lane = tid & 31;
    const int w = tid >> 5;                  // wave 0..3

    const int bid = blockIdx.x;
    const int qt = bid & 31;                 // S/64 = 32 query tiles
    const int h = (bid >> 5) & 15;
    const int b = bid >> 9;
    const int qbase = qt * 64;

    const int r = lane & 15;
    const int khB = (lane & 16) ? 16 : 0;

    // ---- load Q fragments (held for whole kernel): 16 rows x 64 dk ----
    FragAB qf[2];
    {
        const int srow = qbase + w * 16 + r;
        const size_t base = (size_t)(b * Sc + srow) * Dc + h * DKc;
#pragma unroll
        for (int f = 0; f < 2; ++f)
#pragma unroll
            for (int j = 0; j < 8; ++j)
                qf[f].u[j] =
                    *(const unsigned int*)&Q[base + f * 32 + a_k(j, lane)];
    }

    f32x8 acc[4];
    const f32x8 zf = {0.f, 0.f, 0.f, 0.f, 0.f, 0.f, 0.f, 0.f};
#pragma unroll
    for (int t = 0; t < 4; ++t) acc[t] = zf;
    float rmax[8], rsum[8];
#pragma unroll
    for (int i = 0; i < 8; ++i) { rmax[i] = -1.0e30f; rsum[i] = 0.f; }

    // ---- tile staging: K direct (async-capable), V transposed (manual) ----
    auto stage = [&](int kt, int buf) {
        const int key0 = kt * 64;
        unsigned short* Kt_ = Kt[buf];
        unsigned short* Vt_ = Vt[buf];
#pragma unroll
        for (int i = 0; i < 4; ++i) {        // 512 chunks of 8 elems
            int c = tid + i * 128;
            int row = c >> 3;
            int kc = (c & 7) * 8;
            size_t g = (size_t)(b * Sc + key0 + row) * Dc + h * DKc + kc;
            copy16_to_lds(&K[g], &Kt_[row * PK + kc]);
            uint4 dv4 = *(const uint4*)&V[g];
            const unsigned short* e = (const unsigned short*)&dv4;
#pragma unroll
            for (int q = 0; q < 8; ++q) Vt_[(kc + q) * PK + row] = e[q];
        }
    };

    stage(0, 0);                             // prologue: tile 0 in flight
    constexpr int NT = Sc / 64;              // 32 key tiles

    for (int kt = 0; kt < NT; ++kt) {
        const int cur = kt & 1;
        if (kt + 1 < NT) {
            stage(kt + 1, cur ^ 1);          // issue next tile (other buffer)
            WAIT_ASYNC(4);                   // tile kt's 4 async ops done
        } else {
            WAIT_ASYNC(0);
        }
        __syncthreads();                     // publish tile kt block-wide

        const unsigned short* Kt_ = Kt[cur];
        const unsigned short* Vt_ = Vt[cur];

        // ---- scores S = Q @ K^T (reduction over dk) ----
        f32x8 sc[4];
#pragma unroll
        for (int ni = 0; ni < 4; ++ni) {
            sc[ni] = zf;
            const unsigned short* krow = &Kt_[(ni * 16 + r) * PK];
#pragma unroll
            for (int f = 0; f < 2; ++f) {
                FragAB kf;
#pragma unroll
                for (int j = 0; j < 8; ++j)
                    kf.u[j] =
                        *(const unsigned int*)&krow[f * 32 + a_k(j, lane)];
                sc[ni] = __builtin_amdgcn_wmma_f32_16x16x32_bf16(
                    false, qf[f].v, false, kf.v, (short)0, sc[ni], false,
                    false);
            }
            sc[ni] *= SCALE;
        }

        // ---- online softmax per row; write P (bf16) to per-wave LDS ----
        const int Mofs = (lane & 16) ? 8 : 0;
        unsigned short* pw = &Pt[w * 16 * PK];
#pragma unroll
        for (int rr = 0; rr < 8; ++rr) {
            float m = fmaxf(fmaxf(sc[0][rr], sc[1][rr]),
                            fmaxf(sc[2][rr], sc[3][rr]));
            m = fmaxf(m, __shfl_xor(m, 1, 32));
            m = fmaxf(m, __shfl_xor(m, 2, 32));
            m = fmaxf(m, __shfl_xor(m, 4, 32));
            m = fmaxf(m, __shfl_xor(m, 8, 32));
            float mn = fmaxf(rmax[rr], m);
            float corr = __expf(rmax[rr] - mn);
            rmax[rr] = mn;
            float p0 = __expf(sc[0][rr] - mn);
            float p1 = __expf(sc[1][rr] - mn);
            float p2 = __expf(sc[2][rr] - mn);
            float p3 = __expf(sc[3][rr] - mn);
            float ls = p0 + p1 + p2 + p3;
            ls += __shfl_xor(ls, 1, 32);
            ls += __shfl_xor(ls, 2, 32);
            ls += __shfl_xor(ls, 4, 32);
            ls += __shfl_xor(ls, 8, 32);
            rsum[rr] = rsum[rr] * corr + ls;
#pragma unroll
            for (int t = 0; t < 4; ++t) acc[t][rr] *= corr;
            unsigned short* prow = &pw[(rr + Mofs) * PK + r];
            prow[0]  = f2bf(p0);
            prow[16] = f2bf(p1);
            prow[32] = f2bf(p2);
            prow[48] = f2bf(p3);
        }

        // ---- O += P @ V (reduction over 64 keys; V transposed in LDS) ----
        FragAB pf[2];
#pragma unroll
        for (int f = 0; f < 2; ++f) {
            const unsigned short* prow = &pw[r * PK + f * 32];
#pragma unroll
            for (int j = 0; j < 8; ++j)
                pf[f].u[j] = *(const unsigned int*)&prow[a_k(j, lane)];
        }
#pragma unroll
        for (int t = 0; t < 4; ++t) {
            const unsigned short* vrow = &Vt_[(t * 16 + r) * PK];
#pragma unroll
            for (int f = 0; f < 2; ++f) {
                FragAB vf;
#pragma unroll
                for (int j = 0; j < 8; ++j)
                    vf.u[j] =
                        *(const unsigned int*)&vrow[f * 32 + a_k(j, lane)];
                acc[t] = __builtin_amdgcn_wmma_f32_16x16x32_bf16(
                    false, pf[f].v, false, vf.v, (short)0, acc[t], false,
                    false);
            }
        }
        __syncthreads();                     // all reads done; buffer reusable
    }

    // ---- normalize and store ctx[b, s, h*64+dk] as bf16 ----
    const int Mofs = (lane & 16) ? 8 : 0;
#pragma unroll
    for (int t = 0; t < 4; ++t) {
        const int dk = t * 16 + r;
#pragma unroll
        for (int rr = 0; rr < 8; ++rr) {
            const int srow = qbase + w * 16 + rr + Mofs;
            float v = acc[t][rr] / rsum[rr];
            ctx[(size_t)(b * Sc + srow) * Dc + h * DKc + dk] = f2bf(v);
        }
    }
}

// ---------------------------------------------------------------------------
// Host-side orchestration
// ---------------------------------------------------------------------------
extern "C" void kernel_launch(void* const* d_in, const int* in_sizes, int n_in,
                              void* d_out, int out_size, void* d_ws,
                              size_t ws_size, hipStream_t stream) {
    (void)in_sizes; (void)n_in; (void)out_size; (void)ws_size;
    const float* x  = (const float*)d_in[0];
    const float* Wq = (const float*)d_in[1];
    const float* bq = (const float*)d_in[2];
    const float* Wk = (const float*)d_in[3];
    const float* bk = (const float*)d_in[4];
    const float* Wv = (const float*)d_in[5];
    const float* bv = (const float*)d_in[6];
    const float* Wo = (const float*)d_in[7];
    const float* bo = (const float*)d_in[8];

    const size_t XB = (size_t)Mrows * Dc * sizeof(unsigned short); // 16 MiB
    const size_t WB = (size_t)Dc * Dc * sizeof(unsigned short);    //  2 MiB
    char* ws = (char*)d_ws;
    unsigned short* xb  = (unsigned short*)(ws);
    unsigned short* wqb = (unsigned short*)(ws + XB);
    unsigned short* wkb = (unsigned short*)(ws + XB + WB);
    unsigned short* wvb = (unsigned short*)(ws + XB + 2 * WB);
    unsigned short* wob = (unsigned short*)(ws + XB + 3 * WB);
    unsigned short* qb  = (unsigned short*)(ws + XB + 4 * WB);
    unsigned short* kb  = (unsigned short*)(ws + 2 * XB + 4 * WB);
    unsigned short* vb  = (unsigned short*)(ws + 3 * XB + 4 * WB);
    unsigned short* cx  = (unsigned short*)(ws + 4 * XB + 4 * WB);

    // fp32 -> bf16
    cvt_f32_bf16<<<2048, 256, 0, stream>>>(x,  xb,  Mrows * Dc);
    cvt_f32_bf16<<<512,  256, 0, stream>>>(Wq, wqb, Dc * Dc);
    cvt_f32_bf16<<<512,  256, 0, stream>>>(Wk, wkb, Dc * Dc);
    cvt_f32_bf16<<<512,  256, 0, stream>>>(Wv, wvb, Dc * Dc);
    cvt_f32_bf16<<<512,  256, 0, stream>>>(Wo, wob, Dc * Dc);

    // QKV projections (bf16 out)
    const int gemmGrid = (Mrows / 128) * (Dc / 128);  // 512
    gemm_bias_kernel<true><<<gemmGrid, 256, 0, stream>>>(xb, wqb, bq, qb);
    gemm_bias_kernel<true><<<gemmGrid, 256, 0, stream>>>(xb, wkb, bk, kb);
    gemm_bias_kernel<true><<<gemmGrid, 256, 0, stream>>>(xb, wvb, bv, vb);

    // flash attention: B*H*(S/64) = 2048 blocks
    flash_attn_kernel<<<Bc * Hc * (Sc / 64), 128, 0, stream>>>(qb, kb, vb, cx);

    // output projection (fp32 out)
    gemm_bias_kernel<false><<<gemmGrid, 256, 0, stream>>>(cx, wob, bo,
                                                          (float*)d_out);
}
